// HRNetVit3_22539988369896
// MI455X (gfx1250) — matrix operand order repaired and checked by reading, not compiled, not run on hardware
//
#include <hip/hip_runtime.h>

// ---------------------------------------------------------------------------
// CDNA5 / gfx1250 implementation of the 3-branch ViT reference.
// All big GEMMs run through v_wmma_f32_16x16x32_bf16 (wave32 WMMA).
// Round 2: templated GEMM epilogues (no runtime branches) + software-
// pipelined global->LDS staging so loads overlap the WMMA pipe.
// ---------------------------------------------------------------------------

typedef __attribute__((ext_vector_type(16))) __bf16 v16bf;
typedef __attribute__((ext_vector_type(8)))  float  v8f;

union FragAB { v16bf v; uint4 q[2]; };

#define D_MODEL 512
#define SEQ_PAD 256
#define NBATCH  128
#define NTOK    (NBATCH * SEQ_PAD)   // 32768
#define HD      128
#define NHEADS  4
#define SEQ_VAL 251                  // 250 tokens + cls

__device__ __forceinline__ unsigned short f2bf(float f) {
  unsigned int u = __float_as_uint(f);
  return (unsigned short)((u + 0x7FFFu + ((u >> 16) & 1u)) >> 16);  // RNE
}
__device__ __forceinline__ float gelu_exact(float v) {
  return 0.5f * v * (1.0f + erff(v * 0.70710678118654752f));
}

// column gather lists: _cols(idx) = [2i for idx] + [2i+1 for idx]
__constant__ int COLS_L[12] = {8,10,12,22,24,26, 9,11,13,23,25,27};
__constant__ int COLS_R[12] = {2, 4, 6,28,30,32, 3, 5, 7,29,31,33};
__constant__ int COLS_M[10] = {0,14,16,18,20, 1,15,17,19,21};

// ---------------------------------------------------------------------------
// Embedding: h[tok][d] = (i==0 ? cls : x_cols @ W + b) + pos[i];  pad rows = 0
// ---------------------------------------------------------------------------
__global__ void __launch_bounds__(256) embed_kernel(
    const float* __restrict__ x, const float* __restrict__ w,
    const float* __restrict__ bias, const float* __restrict__ pos,
    const float* __restrict__ cls, float* __restrict__ h, int br, int ncols)
{
  const int tok = blockIdx.x;
  const int b = tok >> 8, i = tok & 255;
  const int d0 = threadIdx.x * 2;
  float* out = h + (size_t)tok * D_MODEL + d0;
  if (i >= SEQ_VAL) { out[0] = 0.0f; out[1] = 0.0f; return; }
  const int* cols = (br == 0) ? COLS_L : (br == 1) ? COLS_R : COLS_M;
  float a0, a1;
  if (i == 0) {
    a0 = cls[d0]; a1 = cls[d0 + 1];
  } else {
    a0 = bias[d0]; a1 = bias[d0 + 1];
    const float* xr = x + ((size_t)b * 250 + (i - 1)) * 34;
    for (int j = 0; j < ncols; ++j) {
      const float xv = xr[cols[j]];
      a0 += xv * w[j * D_MODEL + d0];
      a1 += xv * w[j * D_MODEL + d0 + 1];
    }
  }
  a0 += pos[(size_t)i * D_MODEL + d0];
  a1 += pos[(size_t)i * D_MODEL + d0 + 1];
  out[0] = a0; out[1] = a1;
}

// ---------------------------------------------------------------------------
// LayerNorm over D=512, output bf16.  grid = NTOK blocks x 128 threads.
// ---------------------------------------------------------------------------
__global__ void __launch_bounds__(128) ln512(
    const float* __restrict__ H, const float* __restrict__ G,
    const float* __restrict__ Bt, unsigned short* __restrict__ Z)
{
  __shared__ float red[128];
  const int row = blockIdx.x, tid = threadIdx.x;
  const float* hr = H + (size_t)row * D_MODEL;
  float s = 0.0f, s2 = 0.0f;
  #pragma unroll
  for (int j = tid; j < D_MODEL; j += 128) { const float v = hr[j]; s += v; s2 += v * v; }
  red[tid] = s; __syncthreads();
  for (int o = 64; o > 0; o >>= 1) { if (tid < o) red[tid] += red[tid + o]; __syncthreads(); }
  const float mean = red[0] * (1.0f / 512.0f);
  __syncthreads();
  red[tid] = s2; __syncthreads();
  for (int o = 64; o > 0; o >>= 1) { if (tid < o) red[tid] += red[tid + o]; __syncthreads(); }
  const float var = red[0] * (1.0f / 512.0f) - mean * mean;
  const float inv = rsqrtf(var + 1e-5f);
  unsigned short* zr = Z + (size_t)row * D_MODEL;
  #pragma unroll
  for (int j = tid; j < D_MODEL; j += 128)
    zr[j] = f2bf((hr[j] - mean) * inv * G[j] + Bt[j]);
}

// ---------------------------------------------------------------------------
// Tiled GEMM: C[32768,512] = A(bf16) @ W(f32->bf16) [512,512]
// MODE 0: outB = bf16(result)
// MODE 1: outB = bf16(gelu(result + bias))
// MODE 2: outF += result + bias      (residual)
// MODE 3: outF += result             (residual, no bias)
// Block 128x128, 8 waves, wave tile 32(M)x64(N) = 2x4 WMMA tiles, K-step 32.
// Global fetch of chunk k+1 is register-staged while chunk k runs WMMA.
// ---------------------------------------------------------------------------
template<int MODE>
__global__ void __launch_bounds__(256) gemm512(
    const unsigned short* __restrict__ A, const float* __restrict__ W,
    const float* __restrict__ bias, float* __restrict__ outF,
    unsigned short* __restrict__ outB)
{
  __shared__ __align__(16) unsigned short As[128 * 32];  // [m][k]
  __shared__ __align__(16) unsigned short Bs[128 * 32];  // [n][k] (transposed)
  const int tid = threadIdx.x;
  const int lane = tid & 31;
  const int wv = tid >> 5;
  const int m0 = blockIdx.x * 128;
  const int n0 = blockIdx.y * 128;
  const int wm = (wv & 3) * 32;
  const int wn = (wv >> 2) * 64;

  v8f acc[2][4];
  #pragma unroll
  for (int i = 0; i < 2; ++i)
    #pragma unroll
    for (int j = 0; j < 4; ++j)
      acc[i][j] = (v8f){0.f,0.f,0.f,0.f,0.f,0.f,0.f,0.f};

  const int ar = tid >> 1, ac = (tid & 1) * 16;   // A loader: 128 rows, 2 thr/row
  const int wr = tid >> 3, wc = (tid & 7) * 16;   // W loader: 32 rows, 8 thr/row
  const int kb  = (lane < 16) ? 0 : 8;
  const int kb2 = (lane < 16) ? 0 : 16;
  const unsigned short* Aptr = A + (size_t)(m0 + ar) * D_MODEL + ac;
  const float*          Wptr = W + (size_t)wr * D_MODEL + n0 + wc;

  // register-staged tiles (software pipeline)
  uint4  aR[2];
  float4 wR[4];
  aR[0] = ((const uint4*)Aptr)[0];
  aR[1] = ((const uint4*)Aptr)[1];
  #pragma unroll
  for (int j = 0; j < 4; ++j) wR[j] = ((const float4*)Wptr)[j];

  for (int k0 = 0; k0 < D_MODEL; k0 += 32) {
    { // commit staged registers to LDS
      uint4* d = (uint4*)(As + ar * 32 + ac);
      d[0] = aR[0]; d[1] = aR[1];
      const float* wf = (const float*)wR;
      #pragma unroll
      for (int j = 0; j < 16; ++j)
        Bs[(wc + j) * 32 + wr] = f2bf(wf[j]);
    }
    __syncthreads();
    if (k0 + 32 < D_MODEL) {  // stage next chunk while WMMAs run
      const unsigned short* an = Aptr + k0 + 32;
      const float*          wn2 = Wptr + (size_t)(k0 + 32) * D_MODEL;
      aR[0] = ((const uint4*)an)[0];
      aR[1] = ((const uint4*)an)[1];
      #pragma unroll
      for (int j = 0; j < 4; ++j) wR[j] = ((const float4*)wn2)[j];
    }
    FragAB a[2], b[4];
    #pragma unroll
    for (int i = 0; i < 2; ++i) {
      const unsigned short* p = As + (wm + 16 * i + (lane & 15)) * 32;
      a[i].q[0] = *(const uint4*)(p + kb);
      a[i].q[1] = *(const uint4*)(p + kb + 16);
    }
    #pragma unroll
    for (int j = 0; j < 4; ++j) {
      const unsigned short* p = Bs + (wn + 16 * j + (lane & 15)) * 32;
      b[j].q[0] = *(const uint4*)(p + kb);
      b[j].q[1] = *(const uint4*)(p + kb + 16);
    }
    #pragma unroll
    for (int i = 0; i < 2; ++i)
      #pragma unroll
      for (int j = 0; j < 4; ++j)
        acc[i][j] = __builtin_amdgcn_wmma_f32_16x16x32_bf16(
            false, a[i].v, false, b[j].v, (short)0, acc[i][j], false, false);
    __syncthreads();
  }

  const int rb = (lane < 16) ? 0 : 8;
  #pragma unroll
  for (int i = 0; i < 2; ++i) {
    #pragma unroll
    for (int j = 0; j < 4; ++j) {
      const int col = n0 + wn + 16 * j + (lane & 15);
      const float bv = (MODE == 1 || MODE == 2) ? bias[col] : 0.0f;
      #pragma unroll
      for (int r = 0; r < 8; ++r) {
        const int row = m0 + wm + 16 * i + rb + r;
        const size_t idx = (size_t)row * D_MODEL + col;
        float vvv = acc[i][j][r] + bv;
        if (MODE == 0) {
          outB[idx] = f2bf(vvv);
        } else if (MODE == 1) {
          outB[idx] = f2bf(gelu_exact(vvv));
        } else {
          outF[idx] += vvv;
        }
      }
    }
  }
}

// ---------------------------------------------------------------------------
// Fused attention: per (b, head, 16-query tile): S = QK^T/sqrt(hd),
// masked softmax (251 valid keys), O = P V.  128 threads = 4 waves.
// ---------------------------------------------------------------------------
__global__ void __launch_bounds__(128) attn512(
    const unsigned short* __restrict__ Q, const unsigned short* __restrict__ K,
    const unsigned short* __restrict__ V, unsigned short* __restrict__ O)
{
  __shared__ __align__(16) unsigned short q_s[16 * 128];   // [q][d]
  __shared__ __align__(16) unsigned short kv_s[64 * 128];  // k:[key][d] / v:[d][key]
  __shared__ __align__(16) float          lg[16 * 256];    // logits
  __shared__ __align__(16) unsigned short at[16 * 256];    // probs bf16
  __shared__ float sm[16][8];                              // softmax reductions
  const int tid = threadIdx.x, lane = tid & 31, wv = tid >> 5;
  const int idx = blockIdx.x;
  const int qt = idx & 15, hh = (idx >> 4) & 3, bb = idx >> 6;
  const size_t base = (size_t)bb * SEQ_PAD * D_MODEL + hh * HD;
  const int q0 = qt * 16;
  const int kb  = (lane < 16) ? 0 : 8;
  const int kb2 = (lane < 16) ? 0 : 16;

  { // stage Q tile [16][128]
    const int r = tid >> 3, c = (tid & 7) * 16;
    const uint4* s = (const uint4*)(Q + base + (size_t)(q0 + r) * D_MODEL + c);
    uint4* d = (uint4*)(q_s + r * 128 + c);
    d[0] = s[0]; d[1] = s[1];
  }

  // ---- logits = Q K^T * scale, key stages of 64 ----
  for (int s4 = 0; s4 < 4; ++s4) {
    __syncthreads();
    { // stage K [64 keys][128 d]
      const int r = tid >> 1, c = (tid & 1) * 64;
      const uint4* s = (const uint4*)(K + base + (size_t)(s4 * 64 + r) * D_MODEL + c);
      uint4* d = (uint4*)(kv_s + r * 128 + c);
      #pragma unroll
      for (int j = 0; j < 8; ++j) d[j] = s[j];
    }
    __syncthreads();
    v8f c8 = (v8f){0.f,0.f,0.f,0.f,0.f,0.f,0.f,0.f};
    const unsigned short* ap = q_s + (lane & 15) * 128;
    const unsigned short* bp = kv_s + (wv * 16 + (lane & 15)) * 128;
    #pragma unroll
    for (int d0 = 0; d0 < HD; d0 += 32) {
      FragAB a, b;
      a.q[0] = *(const uint4*)(ap + d0 + kb);
      a.q[1] = *(const uint4*)(ap + d0 + kb + 16);
      b.q[0] = *(const uint4*)(bp + d0 + kb);
      b.q[1] = *(const uint4*)(bp + d0 + kb + 16);
      c8 = __builtin_amdgcn_wmma_f32_16x16x32_bf16(
          false, a.v, false, b.v, (short)0, c8, false, false);
    }
    const int col = s4 * 64 + wv * 16 + (lane & 15);
    const int rb = (lane < 16) ? 0 : 8;
    #pragma unroll
    for (int r = 0; r < 8; ++r)
      lg[(rb + r) * 256 + col] = c8[r] * 0.08838834764831845f;  // 1/sqrt(128)
  }
  __syncthreads();

  // ---- masked softmax over 251 keys (8 threads per row), probs -> bf16 ----
  {
    const int r = tid >> 3, sg = tid & 7;   // row 0..15, segment 0..7 (32 cols)
    float* row = lg + r * 256;
    float mx = -3.0e38f;
    for (int j = sg * 32; j < sg * 32 + 32; ++j)
      if (j < SEQ_VAL) mx = fmaxf(mx, row[j]);
    sm[r][sg] = mx; __syncthreads();
    if (sg == 0) {
      float m2 = sm[r][0];
      #pragma unroll
      for (int t = 1; t < 8; ++t) m2 = fmaxf(m2, sm[r][t]);
      sm[r][0] = m2;
    }
    __syncthreads();
    const float rowmax = sm[r][0];
    float s = 0.0f;
    for (int j = sg * 32; j < sg * 32 + 32; ++j)
      if (j < SEQ_VAL) { const float e = expf(row[j] - rowmax); row[j] = e; s += e; }
    __syncthreads();               // rowmax consumed before sm reuse
    sm[r][sg] = s; __syncthreads();
    if (sg == 0) {
      float t2 = 0.0f;
      #pragma unroll
      for (int t = 0; t < 8; ++t) t2 += sm[r][t];
      sm[r][0] = t2;
    }
    __syncthreads();
    const float inv = 1.0f / sm[r][0];
    unsigned short* arow = at + r * 256;
    for (int j = sg * 32; j < sg * 32 + 32; ++j)
      arow[j] = (j < SEQ_VAL) ? f2bf(row[j] * inv) : (unsigned short)0;
  }

  // ---- O = P V, V staged transposed [d][key] ----
  v8f o8[2];
  o8[0] = (v8f){0.f,0.f,0.f,0.f,0.f,0.f,0.f,0.f};
  o8[1] = (v8f){0.f,0.f,0.f,0.f,0.f,0.f,0.f,0.f};
  for (int s4 = 0; s4 < 4; ++s4) {
    __syncthreads();
    {
      const int kl = tid >> 1, dd0 = (tid & 1) * 64;
      const unsigned short* s = V + base + (size_t)(s4 * 64 + kl) * D_MODEL + dd0;
      for (int j = 0; j < 64; ++j) kv_s[(dd0 + j) * 64 + kl] = s[j];
    }
    __syncthreads();
    #pragma unroll
    for (int t = 0; t < 2; ++t) {
      const int dt = wv + 4 * t;
      #pragma unroll
      for (int c2 = 0; c2 < 2; ++c2) {
        FragAB a, b;
        const unsigned short* ap = at + (lane & 15) * 256 + s4 * 64 + c2 * 32;
        a.q[0] = *(const uint4*)(ap + kb);
        a.q[1] = *(const uint4*)(ap + kb + 16);
        const unsigned short* bp = kv_s + (dt * 16 + (lane & 15)) * 64 + c2 * 32;
        b.q[0] = *(const uint4*)(bp + kb);
        b.q[1] = *(const uint4*)(bp + kb + 16);
        o8[t] = __builtin_amdgcn_wmma_f32_16x16x32_bf16(
            false, a.v, false, b.v, (short)0, o8[t], false, false);
      }
    }
  }
  const int rb = (lane < 16) ? 0 : 8;
  #pragma unroll
  for (int t = 0; t < 2; ++t) {
    const int dc = (wv + 4 * t) * 16 + (lane & 15);
    #pragma unroll
    for (int r = 0; r < 8; ++r)
      O[base + (size_t)(q0 + rb + r) * D_MODEL + dc] = f2bf(o8[t][r]);
  }
}

// ---------------------------------------------------------------------------
// CLS gather + tiny head
// ---------------------------------------------------------------------------
__global__ void __launch_bounds__(256) cls_copy(const float* __restrict__ h,
                                               float* __restrict__ cat, int br)
{
  const int gid = blockIdx.x * 256 + threadIdx.x;   // 65536 = 128*512
  const int b = gid >> 9, j = gid & 511;
  cat[(size_t)b * 1536 + br * 512 + j] = h[(size_t)b * SEQ_PAD * D_MODEL + j];
}

__global__ void __launch_bounds__(256) head1(const float* __restrict__ cat,
                                             const float* __restrict__ W,
                                             const float* __restrict__ bias,
                                             float* __restrict__ hid)
{
  const int gid = blockIdx.x * 256 + threadIdx.x;
  const int b = gid >> 9, j = gid & 511;
  float acc = bias[j];
  const float* cr = cat + (size_t)b * 1536;
  for (int i = 0; i < 1536; ++i) acc += cr[i] * W[(size_t)i * 512 + j];
  hid[gid] = gelu_exact(acc);
}

__global__ void __launch_bounds__(128) head2(const float* __restrict__ hid,
                                             const float* __restrict__ W,
                                             const float* __restrict__ bias,
                                             float* __restrict__ out)
{
  const int b = threadIdx.x;
  if (b >= 128) return;
  float acc = bias[0];
  for (int j = 0; j < 512; ++j) acc += hid[(size_t)b * 512 + j] * W[j];
  out[b] = 1.0f / (1.0f + expf(-acc));
}

// ---------------------------------------------------------------------------
// Host orchestration
// ---------------------------------------------------------------------------
extern "C" void kernel_launch(void* const* d_in, const int* in_sizes, int n_in,
                              void* d_out, int out_size, void* d_ws, size_t ws_size,
                              hipStream_t stream) {
  (void)in_sizes; (void)n_in; (void)out_size; (void)ws_size;
  const float* x        = (const float*)d_in[0];
  const float* w_left   = (const float*)d_in[1];
  const float* b_left   = (const float*)d_in[2];
  const float* w_right  = (const float*)d_in[3];
  const float* b_right  = (const float*)d_in[4];
  const float* w_mid    = (const float*)d_in[5];
  const float* b_mid    = (const float*)d_in[6];
  const float* pos_emb  = (const float*)d_in[7];
  const float* cls_tok  = (const float*)d_in[8];
  const float* ln1_g    = (const float*)d_in[9];
  const float* ln1_b    = (const float*)d_in[10];
  const float* ln2_g    = (const float*)d_in[11];
  const float* ln2_b    = (const float*)d_in[12];
  const float* wq       = (const float*)d_in[13];
  const float* wk       = (const float*)d_in[14];
  const float* wvv      = (const float*)d_in[15];
  const float* wo       = (const float*)d_in[16];
  const float* fc1_w    = (const float*)d_in[17];
  const float* fc1_b    = (const float*)d_in[18];
  const float* fc2_w    = (const float*)d_in[19];
  const float* fc2_b    = (const float*)d_in[20];
  const float* out1_w   = (const float*)d_in[21];
  const float* out1_b   = (const float*)d_in[22];
  const float* out2_w   = (const float*)d_in[23];
  const float* out2_b   = (const float*)d_in[24];

  char* ws = (char*)d_ws;
  const size_t MB = (size_t)1 << 20;
  float*          h   = (float*)ws;                               // 64 MB
  unsigned short* z   = (unsigned short*)(ws + 64 * MB);          // 32 MB
  unsigned short* q   = (unsigned short*)(ws + 96 * MB);          // 32 MB (reused as m1)
  unsigned short* k   = (unsigned short*)(ws + 128 * MB);         // 32 MB
  unsigned short* v   = (unsigned short*)(ws + 160 * MB);         // 32 MB
  unsigned short* o   = (unsigned short*)(ws + 192 * MB);         // 32 MB
  float*          cat = (float*)(ws + 224 * MB);                  // 768 KB
  float*          hid = (float*)(ws + 225 * MB);                  // 256 KB

  const dim3 ggrid(NTOK / 128, D_MODEL / 128);   // 256 x 4

  for (int br = 0; br < 3; ++br) {
    const float* wemb = (br == 0) ? w_left : (br == 1) ? w_right : w_mid;
    const float* bemb = (br == 0) ? b_left : (br == 1) ? b_right : b_mid;
    const int ncols = (br == 2) ? 10 : 12;
    embed_kernel<<<NTOK, 256, 0, stream>>>(x, wemb, bemb, pos_emb, cls_tok, h, br, ncols);
    for (int l = 0; l < 4; ++l) {
      const size_t po = (size_t)(br * 4 + l) * 512;
      const size_t wf = (size_t)(br * 4 + l) * 512 * 512;
      ln512<<<NTOK, 128, 0, stream>>>(h, ln1_g + po, ln1_b + po, z);
      gemm512<0><<<ggrid, 256, 0, stream>>>(z, wq  + wf, nullptr, nullptr, q);
      gemm512<0><<<ggrid, 256, 0, stream>>>(z, wk  + wf, nullptr, nullptr, k);
      gemm512<0><<<ggrid, 256, 0, stream>>>(z, wvv + wf, nullptr, nullptr, v);
      attn512<<<NBATCH * NHEADS * 16, 128, 0, stream>>>(q, k, v, o);
      gemm512<3><<<ggrid, 256, 0, stream>>>(o, wo + wf, nullptr, h, nullptr);        // h += O@Wo
      ln512<<<NTOK, 128, 0, stream>>>(h, ln2_g + po, ln2_b + po, z);
      gemm512<1><<<ggrid, 256, 0, stream>>>(z, fc1_w + wf, fc1_b + po, nullptr, q);  // m1 = gelu
      gemm512<2><<<ggrid, 256, 0, stream>>>(q, fc2_w + wf, fc2_b + po, h, nullptr);  // h += m1@fc2+b
    }
    cls_copy<<<256, 256, 0, stream>>>(h, cat, br);
  }
  head1<<<256, 256, 0, stream>>>(cat, out1_w, out1_b, hid);
  head2<<<1, 128, 0, stream>>>(hid, out2_w, out2_b, (float*)d_out);
}
